// EncoderLayer_1520418423070
// MI455X (gfx1250) — matrix operand order repaired and checked by reading, not compiled, not run on hardware
//
#include <hip/hip_runtime.h>
#include <hip/hip_bf16.h>

// ---------------------------------------------------------------------------
// Problem constants: B=16, S=1024, D=H=U=512
// ---------------------------------------------------------------------------
static constexpr int BB = 16;
static constexpr int SS = 1024;
static constexpr int DD = 512;
static constexpr int HH = 512;
static constexpr int UU = 512;

typedef __bf16 v16bf __attribute__((ext_vector_type(16)));
typedef __bf16 v8bf  __attribute__((ext_vector_type(8)));
typedef __bf16 v4bf  __attribute__((ext_vector_type(4)));
typedef float  v8f   __attribute__((ext_vector_type(8)));
typedef unsigned int u32x4 __attribute__((ext_vector_type(4)));
typedef int i32x4 __attribute__((ext_vector_type(4)));
typedef int i32x8 __attribute__((ext_vector_type(8)));

#define GLOBAL_AS __attribute__((address_space(1)))
#define LDS_AS    __attribute__((address_space(3)))

// --- CDNA5 feature probes (fallbacks keep the compile safe) ----------------
#if defined(__has_builtin)
#if __has_builtin(__builtin_amdgcn_global_load_async_to_lds_b128) && \
    __has_builtin(__builtin_amdgcn_s_wait_asynccnt)
#define USE_ASYNC_LDS 1
#endif
#if __has_builtin(__builtin_amdgcn_tensor_load_to_lds) && \
    __has_builtin(__builtin_amdgcn_s_wait_tensorcnt)
#define USE_TDM 1
#endif
#endif

// 16-byte global -> LDS copy: async (ASYNCcnt) when available.
// Builtin signature (from clang diagnostic): param0 = addrspace(1) int4*
// (global source), param1 = LDS int4* destination, then offset, cpol.
__device__ inline void async_cp16(void* lds, const void* g) {
#ifdef USE_ASYNC_LDS
  __builtin_amdgcn_global_load_async_to_lds_b128(
      (GLOBAL_AS i32x4*)g, (LDS_AS i32x4*)lds, 0, 0);
#else
  *(uint4*)lds = *(const uint4*)g;
#endif
}
__device__ inline void wait_async_all() {
#ifdef USE_ASYNC_LDS
  __builtin_amdgcn_s_wait_asynccnt(0);
#endif
}

// ---------------------------------------------------------------------------
// WMMA helpers (16x16x32 bf16, f32 accumulate)
// ---------------------------------------------------------------------------
__device__ inline v16bf make16(const __bf16* lo, const __bf16* hi) {
  v8bf a = *(const v8bf*)lo;
  v8bf b = *(const v8bf*)hi;
  return __builtin_shufflevector(a, b, 0,1,2,3,4,5,6,7,8,9,10,11,12,13,14,15);
}
__device__ inline v8f wmma_bf16(v16bf a, v16bf b, v8f c) {
  return __builtin_amdgcn_wmma_f32_16x16x32_bf16(false, a, false, b, (short)0, c,
                                                 false, false);
}

// ---------------------------------------------------------------------------
// Grid-wide two-phase barrier (counters zeroed each launch)
// ---------------------------------------------------------------------------
__device__ inline void grid_sync(int* cnt, int* gen, int nblk) {
  __syncthreads();
  if (threadIdx.x == 0) {
    int g = __atomic_load_n(gen, __ATOMIC_ACQUIRE);
    if (atomicAdd(cnt, 1) == nblk - 1) {
      atomicExch(cnt, 0);
      __threadfence();
      atomicAdd(gen, 1);
    } else {
      while (__atomic_load_n(gen, __ATOMIC_ACQUIRE) == g)
        __builtin_amdgcn_s_sleep(2);
    }
  }
  __syncthreads();
}
__global__ void init_barrier_kernel(int* p) { p[0] = 0; p[1] = 0; }

// ---------------------------------------------------------------------------
// One-time f32 -> bf16 conversion (flat) and convert+transpose for weights.
// ---------------------------------------------------------------------------
__global__ void __launch_bounds__(256)
cvt_bf16_kernel(const float* __restrict__ in, __bf16* __restrict__ out,
                long long n4) {
  long long i = ((long long)blockIdx.x * 256 + threadIdx.x) * 4;
  if (i >= n4 * 4) return;
  float4 t = *(const float4*)&in[i];
  v4bf o; o[0] = (__bf16)t.x; o[1] = (__bf16)t.y; o[2] = (__bf16)t.z; o[3] = (__bf16)t.w;
  *(v4bf*)&out[i] = o;
}

// in: f32 [K][N] row-major  ->  out: bf16 [N][K] row-major
__global__ void __launch_bounds__(256)
cvtT_bf16_kernel(const float* __restrict__ in, __bf16* __restrict__ out,
                 int K, int N) {
  __shared__ float tile[32][33];
  int k0 = blockIdx.y * 32, n0 = blockIdx.x * 32;
  int tx = threadIdx.x, ty = threadIdx.y;   // block (32, 8)
  for (int i = 0; i < 32; i += 8)
    tile[ty + i][tx] = in[(size_t)(k0 + ty + i) * N + n0 + tx];
  __syncthreads();
  for (int i = 0; i < 32; i += 8)
    out[(size_t)(n0 + ty + i) * K + k0 + tx] = (__bf16)tile[tx][ty + i];
}

// ---------------------------------------------------------------------------
// Tiled WMMA GEMM, bf16 A/B, f32 accumulate, TC output (+bias).
// A [M][K] (lda). B_NK: B stored [N][K] (ldb = K-stride, pure async copy);
// else B stored [K][N] (ldb, transpose-on-store). Double-buffered LDS tiles,
// one barrier per BK=32 step; async copies overlap WMMA of previous tile.
// ---------------------------------------------------------------------------
template <typename TC, bool B_NK>
__global__ void __launch_bounds__(256)
gemm_wmma(const __bf16* __restrict__ A, const __bf16* __restrict__ Bm,
          TC* __restrict__ C, int K, int lda, int ldb, int ldc,
          long long sA, long long sB, long long sC,
          const float* __restrict__ bias) {
  __shared__ __bf16 lds_a[2][128 * 32];
  __shared__ __bf16 lds_b[2][64 * 32];
  A  += (long long)blockIdx.z * sA;
  Bm += (long long)blockIdx.z * sB;
  C  += (long long)blockIdx.z * sC;
  const int tid = threadIdx.x;
  const int m0 = blockIdx.x * 128, n0 = blockIdx.y * 64;
  const int w = tid >> 5, lane = tid & 31;
  const int wm = w & 3, wn = w >> 2;
  const int mlane = lane & 15, half = lane >> 4;

  auto stageA = [&](int buf, int k0) {
    for (int i = 0; i < 2; i++) {
      int c = tid + i * 256, row = c >> 2, kk8 = (c & 3) * 8;
      async_cp16(&lds_a[buf][row * 32 + kk8],
                 &A[(size_t)(m0 + row) * lda + k0 + kk8]);
    }
  };
  auto stageB = [&](int buf, int k0) {
    if constexpr (B_NK) {
      int row = tid >> 2, kk8 = (tid & 3) * 8;
      async_cp16(&lds_b[buf][row * 32 + kk8],
                 &Bm[(size_t)(n0 + row) * ldb + k0 + kk8]);
    } else {
      int kk = tid >> 3, nn8 = (tid & 7) * 8;
      uint4 r = *(const uint4*)&Bm[(size_t)(k0 + kk) * ldb + n0 + nn8];
      const __bf16* e = (const __bf16*)&r;
      for (int q = 0; q < 8; q++) lds_b[buf][(nn8 + q) * 32 + kk] = e[q];
    }
  };

  v8f acc[2][2] = {};
  int buf = 0;
  stageA(0, 0); stageB(0, 0);
  wait_async_all();
  __syncthreads();
  for (int k0 = 0; k0 < K; k0 += 32) {
    if (k0 + 32 < K) { stageA(buf ^ 1, k0 + 32); stageB(buf ^ 1, k0 + 32); }
    v16bf af[2], bf[2];
    for (int i = 0; i < 2; i++) {
      const __bf16* ar = &lds_a[buf][(wm * 32 + i * 16 + mlane) * 32];
      af[i] = make16(ar + 8 * half, ar + 16 + 8 * half);
      const __bf16* br = &lds_b[buf][(wn * 32 + i * 16 + mlane) * 32];
      bf[i] = make16(br + 8 * half, br + 16 + 8 * half);
    }
    for (int i = 0; i < 2; i++)
      for (int j = 0; j < 2; j++)
        acc[i][j] = wmma_bf16(af[i], bf[j], acc[i][j]);
    buf ^= 1;
    if (k0 + 32 < K) { wait_async_all(); __syncthreads(); }
  }
  for (int i = 0; i < 2; i++)
    for (int j = 0; j < 2; j++) {
      int nc = n0 + wn * 32 + j * 16 + mlane;
      float bb = bias ? bias[nc] : 0.f;
      for (int r = 0; r < 8; r++) {
        int mr = m0 + wm * 32 + i * 16 + r + 8 * half;
        C[(size_t)mr * ldc + nc] = (TC)(acc[i][j][r] + bb);
      }
    }
}

// ---------------------------------------------------------------------------
// Attention: per (batch, 32-query stripe): resident 32x512 Q in LDS,
// double-buffered 64x512 K panels (async), full 32x1024 f32 score stripe in
// LDS, exact softmax, P written bf16. LDS total = 296 KB (< 320 KB WGP).
// ---------------------------------------------------------------------------
__global__ void __launch_bounds__(256)
attn_softmax_kernel(const __bf16* __restrict__ Q, const __bf16* __restrict__ Kmat,
                    __bf16* __restrict__ P) {
  extern __shared__ char smem[];
  __bf16* lds_q = (__bf16*)smem;                              // 32*512
  __bf16* lds_k = (__bf16*)(smem + 32 * 512 * 2);             // 2 * 64*512
  float*  sbuf  = (float*)(smem + 32 * 512 * 2 + 2 * 64 * 512 * 2); // 32*1024
  float*  red   = sbuf + 32 * 1024;                           // 32*8
  const int b = blockIdx.y, q0 = blockIdx.x * 32;
  const int tid = threadIdx.x, w = tid >> 5, lane = tid & 31;
  const int wm = w & 1, wn = w >> 1;          // 2 waves M x 4 waves N
  const int mlane = lane & 15, half = lane >> 4;
  const __bf16* Qb = Q + ((size_t)b * SS + q0) * HH;
  const __bf16* Kb = Kmat + (size_t)b * SS * HH;
  const float scale = 0.044194173824159216f;  // 1/sqrt(512)

  auto stageK = [&](int buf, int kt) {
    for (int i = 0; i < 16; i++) {
      int c = tid + i * 256, row = c >> 6, kk8 = (c & 63) * 8;
      async_cp16(&lds_k[buf * 64 * 512 + row * 512 + kk8],
                 &Kb[(size_t)(kt * 64 + row) * HH + kk8]);
    }
  };
  // resident Q stripe
  for (int i = 0; i < 8; i++) {
    int c = tid + i * 256, row = c >> 6, kk8 = (c & 63) * 8;
    async_cp16(&lds_q[row * 512 + kk8], &Qb[(size_t)row * HH + kk8]);
  }
  stageK(0, 0);
  wait_async_all();
  __syncthreads();

  int buf = 0;
  for (int kt = 0; kt < SS / 64; kt++) {
    if (kt + 1 < SS / 64) stageK(buf ^ 1, kt + 1);
    v8f acc = {};
    const __bf16* qrow = &lds_q[(wm * 16 + mlane) * 512];
    const __bf16* krow = &lds_k[buf * 64 * 512 + (wn * 16 + mlane) * 512];
    for (int k0 = 0; k0 < HH; k0 += 32) {
      v16bf a = make16(qrow + k0 + 8 * half, qrow + k0 + 16 + 8 * half);
      v16bf kk = make16(krow + k0 + 8 * half, krow + k0 + 16 + 8 * half);
      acc = wmma_bf16(a, kk, acc);
    }
    for (int r = 0; r < 8; r++) {
      int m = wm * 16 + r + 8 * half;
      int col = kt * 64 + wn * 16 + mlane;
      sbuf[m * SS + col] = acc[r] * scale;
    }
    buf ^= 1;
    if (kt + 1 < SS / 64) { wait_async_all(); __syncthreads(); }
  }
  __syncthreads();

  // exact softmax: 8 threads per row, 128 cols each
  const int row = tid >> 3, part = tid & 7;
  float mx = -3.4e38f;
  for (int i = 0; i < 128; i++) mx = fmaxf(mx, sbuf[row * SS + part * 128 + i]);
  red[row * 8 + part] = mx;
  __syncthreads();
  mx = red[row * 8];
  for (int i = 1; i < 8; i++) mx = fmaxf(mx, red[row * 8 + i]);
  __syncthreads();
  float s = 0.f;
  for (int i = 0; i < 128; i++) {
    float e = __expf(sbuf[row * SS + part * 128 + i] - mx);
    sbuf[row * SS + part * 128 + i] = e;
    s += e;
  }
  red[row * 8 + part] = s;
  __syncthreads();
  s = 0.f;
  for (int i = 0; i < 8; i++) s += red[row * 8 + i];
  const float inv = 1.f / s;
  __bf16* Pr = P + ((size_t)b * SS + q0 + row) * SS;
  for (int i = 0; i < 128; i++)
    Pr[part * 128 + i] = (__bf16)(sbuf[row * SS + part * 128 + i] * inv);
}

// ---------------------------------------------------------------------------
// Residual + LayerNorm, one wave per 512-wide row; optional bf16 aux output.
// ---------------------------------------------------------------------------
__global__ void __launch_bounds__(256)
ln_residual_kernel(const float* __restrict__ x, const float* __restrict__ res,
                   const float* __restrict__ gamma, const float* __restrict__ beta,
                   float* __restrict__ out, __bf16* __restrict__ out_bf, int rows) {
  const int wave = threadIdx.x >> 5, lane = threadIdx.x & 31;
  const int row = blockIdx.x * 8 + wave;
  if (row >= rows) return;
  const float* xr = x + (size_t)row * UU;
  const float* rr = res + (size_t)row * UU;
  float v[16], sum = 0.f;
  for (int i = 0; i < 16; i++) {
    float t = xr[lane + i * 32] + rr[lane + i * 32];
    v[i] = t; sum += t;
  }
  for (int o = 16; o >= 1; o >>= 1) sum += __shfl_xor(sum, o, 32);
  float mu = sum * (1.f / UU);
  float var = 0.f;
  for (int i = 0; i < 16; i++) { float d = v[i] - mu; var += d * d; }
  for (int o = 16; o >= 1; o >>= 1) var += __shfl_xor(var, o, 32);
  float rstd = rsqrtf(var * (1.f / UU) + 1e-3f);
  for (int i = 0; i < 16; i++) {
    int c = lane + i * 32;
    float y = gamma[c] * (v[i] - mu) * rstd + beta[c];
    out[(size_t)row * UU + c] = y;
    if (out_bf) out_bf[(size_t)row * UU + c] = (__bf16)y;
  }
}

// ---------------------------------------------------------------------------
// Persistent LSTM scan: 32 blocks x 128 thr (4 waves = 4 gates). Recurrent
// weight slice resident in LDS for all 1024 steps; per-step 16 KB h-broadcast
// staged by the Tensor Data Mover (TENSORcnt) when available.
// ---------------------------------------------------------------------------
__global__ void __launch_bounds__(128)
lstm_scan_kernel(const float* __restrict__ xp, const float* __restrict__ R,
                 __bf16* __restrict__ hbf, float* __restrict__ hs,
                 int* __restrict__ bar) {
  extern __shared__ char smem[];
  __bf16* lds_r = (__bf16*)smem;                         // 64*512 bf16
  __bf16* lds_h = (__bf16*)(smem + 64 * 512 * 2);        // 16*512 bf16
  float*  zbuf  = (float*)(smem + 64 * 512 * 2 + 16 * 512 * 2);
  float*  cst   = zbuf + 4 * 256;
  const int j = blockIdx.x, tid = threadIdx.x, u0 = j * 16;
  const int w = tid >> 5, lane = tid & 31;
  const int mlane = lane & 15, half = lane >> 4;

  for (int idx = tid; idx < 64 * 512; idx += 128) {
    int col = idx & 63, k = idx >> 6;
    int gc = (col >> 4) * UU + u0 + (col & 15);
    lds_r[col * 512 + k] = (__bf16)R[(size_t)k * (4 * UU) + gc];
  }
  for (int idx = tid; idx < 256; idx += 128) {
    int m = idx >> 4, n = idx & 15;
    hbf[m * UU + u0 + n] = (__bf16)0.f;
    cst[idx] = 0.f;
  }
#ifdef USE_TDM
  // TDM descriptor: 1D tile of 8192 bf16 elements (16x512 h matrix), built
  // once (uniform) — groups 2/3 zero (<=2D tensor).
  u32x4 g0; i32x8 g1; i32x4 gz = {0, 0, 0, 0};
  {
    unsigned lds_off = (unsigned)(uintptr_t)(void*)lds_h;
    unsigned long long ga = (unsigned long long)(uintptr_t)(const void*)hbf;
    g0[0] = 1u;                      // count = 1 valid descriptor
    g0[1] = lds_off;                 // LDS byte address
    g0[2] = (unsigned)ga;            // global addr [31:0]
    g0[3] = (unsigned)((ga >> 32) & 0x01FFFFFFu) | (2u << 30);  // addr[56:32], type=2
    g1[0] = 1 << 16;                 // data_size = 2 bytes
    g1[1] = 0x2000 << 16;            // tensor_dim0 = 8192 (low 16 @ bits 63:48)
    g1[2] = 1 << 16;                 // tensor_dim0 hi = 0 | tensor_dim1 = 1
    g1[3] = 0x2000 << 16;            // tile_dim0 = 8192 @ bits 127:112
    g1[4] = 1;                       // tile_dim1 = 1
    g1[5] = 0x2000;                  // tensor_dim0_stride = 8192
    g1[6] = 0; g1[7] = 0;
  }
#endif
  __threadfence();
  grid_sync(bar, bar + 1, gridDim.x);

  for (int t = 0; t < SS; t++) {
#ifdef USE_TDM
    if (tid < 32) {
#if defined(__clang_major__) && __clang_major__ >= 23
      i32x8 gz8 = {0, 0, 0, 0, 0, 0, 0, 0};
      __builtin_amdgcn_tensor_load_to_lds(g0, g1, gz, gz, gz8, 0);
#else
      __builtin_amdgcn_tensor_load_to_lds(g0, g1, gz, gz, 0);
#endif
      __builtin_amdgcn_s_wait_tensorcnt(0);
    }
    __syncthreads();
#else
    for (int idx = tid; idx < (BB * UU) / 8; idx += 128)
      ((v8bf*)lds_h)[idx] = ((const v8bf*)hbf)[idx];
    __syncthreads();
#endif

    v8f acc = {};
    for (int k0 = 0; k0 < UU; k0 += 32) {
      const __bf16* ar = &lds_h[mlane * 512 + k0];
      v16bf a = make16(ar + 8 * half, ar + 16 + 8 * half);
      const __bf16* br = &lds_r[(w * 16 + mlane) * 512 + k0];
      v16bf bb = make16(br + 8 * half, br + 16 + 8 * half);
      acc = wmma_bf16(a, bb, acc);
    }
    for (int r = 0; r < 8; r++) {
      int m = r + 8 * half, n = mlane;
      float z = acc[r] + xp[((size_t)m * SS + t) * (4 * UU) + w * UU + u0 + n];
      zbuf[w * 256 + m * 16 + n] = z;
    }
    __syncthreads();
    for (int idx = tid; idx < 256; idx += 128) {
      float zi = zbuf[idx], zf = zbuf[256 + idx];
      float zg = zbuf[512 + idx], zo = zbuf[768 + idx];
      float ig = 1.f / (1.f + __expf(-zi));
      float fg = 1.f / (1.f + __expf(-zf));
      float og = 1.f / (1.f + __expf(-zo));
      float c = fg * cst[idx] + ig * fmaxf(zg, 0.f);
      cst[idx] = c;
      float h = og * fmaxf(c, 0.f);
      int m = idx >> 4, n = idx & 15;
      hbf[m * UU + u0 + n] = (__bf16)h;
      hs[((size_t)m * SS + t) * UU + u0 + n] = h;
    }
    __threadfence();
    grid_sync(bar, bar + 1, gridDim.x);
  }
}

// ---------------------------------------------------------------------------
// Host orchestration
// ---------------------------------------------------------------------------
extern "C" void kernel_launch(void* const* d_in, const int* in_sizes, int n_in,
                              void* d_out, int out_size, void* d_ws, size_t ws_size,
                              hipStream_t stream) {
  const float* emb   = (const float*)d_in[0];
  const float* Wq    = (const float*)d_in[1];
  const float* Wk    = (const float*)d_in[2];
  const float* Wv    = (const float*)d_in[3];
  const float* Wo    = (const float*)d_in[4];
  const float* gamma = (const float*)d_in[5];
  const float* beta  = (const float*)d_in[6];
  const float* lk    = (const float*)d_in[7];
  const float* lr    = (const float*)d_in[8];
  const float* lb    = (const float*)d_in[9];
  float* out = (float*)d_out;

  char* ws = (char*)d_ws;
  const int M = BB * SS;                       // 16384
  const size_t F32 = (size_t)M * UU * 4;       // 33.5 MB
  const size_t BF  = (size_t)M * UU * 2;       // 16.8 MB
  size_t off = 0;
  int*    bar  = (int*)(ws + off);     off += 256;
  __bf16* embB = (__bf16*)(ws + off);  off += BF;
  __bf16* WqT  = (__bf16*)(ws + off);  off += (size_t)DD * HH * 2;
  __bf16* WkT  = (__bf16*)(ws + off);  off += (size_t)DD * HH * 2;
  __bf16* WvT  = (__bf16*)(ws + off);  off += (size_t)DD * HH * 2;
  __bf16* WoT  = (__bf16*)(ws + off);  off += (size_t)HH * UU * 2;
  __bf16* lkT  = (__bf16*)(ws + off);  off += (size_t)UU * 4 * UU * 2;
  __bf16* Qb   = (__bf16*)(ws + off);  off += BF;
  __bf16* Kb   = (__bf16*)(ws + off);  off += BF;
  __bf16* Vb   = (__bf16*)(ws + off);  off += BF;
  __bf16* Pm   = (__bf16*)(ws + off);  off += (size_t)BB * SS * SS * 2;
  __bf16* ctxB = (__bf16*)(ws + off);  off += BF;
  float*  ao   = (float*)(ws + off);   off += F32;
  float*  ln1  = (float*)(ws + off);   off += F32;
  __bf16* ln1B = (__bf16*)(ws + off);  off += BF;
  float*  xp   = (float*)(ws + off);   off += (size_t)M * 4 * UU * 4;
  __bf16* hbf  = (__bf16*)(ws + off);  off += (size_t)BB * UU * 2;
  float*  hs   = (float*)(ws + off);   off += F32;

  const dim3 blk256(256), blk128(128), blkT(32, 8);

  init_barrier_kernel<<<1, 1, 0, stream>>>(bar);

  // one-time operand conversions
  cvt_bf16_kernel<<<(M * UU / 4 + 255) / 256, blk256, 0, stream>>>(emb, embB, (long long)M * UU / 4);
  cvtT_bf16_kernel<<<dim3(HH / 32, DD / 32), blkT, 0, stream>>>(Wq, WqT, DD, HH);
  cvtT_bf16_kernel<<<dim3(HH / 32, DD / 32), blkT, 0, stream>>>(Wk, WkT, DD, HH);
  cvtT_bf16_kernel<<<dim3(HH / 32, DD / 32), blkT, 0, stream>>>(Wv, WvT, DD, HH);
  cvtT_bf16_kernel<<<dim3(UU / 32, HH / 32), blkT, 0, stream>>>(Wo, WoT, HH, UU);
  cvtT_bf16_kernel<<<dim3(4 * UU / 32, UU / 32), blkT, 0, stream>>>(lk, lkT, UU, 4 * UU);

  // Q,K,V projections (bf16 out, B pre-transposed)
  dim3 gQKV(M / 128, HH / 64, 1);
  gemm_wmma<__bf16, true><<<gQKV, blk256, 0, stream>>>(embB, WqT, Qb, DD, DD, DD, HH, 0, 0, 0, nullptr);
  gemm_wmma<__bf16, true><<<gQKV, blk256, 0, stream>>>(embB, WkT, Kb, DD, DD, DD, HH, 0, 0, 0, nullptr);
  gemm_wmma<__bf16, true><<<gQKV, blk256, 0, stream>>>(embB, WvT, Vb, DD, DD, DD, HH, 0, 0, 0, nullptr);

  // scores + softmax -> P, 296 KB dynamic LDS per block
  size_t attn_lds = 32 * 512 * 2 + 2 * 64 * 512 * 2 + (size_t)32 * 1024 * 4 + 32 * 8 * 4;
  attn_softmax_kernel<<<dim3(SS / 32, BB), blk256, attn_lds, stream>>>(Qb, Kb, Pm);

  // ctx_b = P_b @ V_b (V is [K][N] activation -> transpose-on-store path)
  dim3 gCTX(SS / 128, HH / 64, BB);
  gemm_wmma<__bf16, false><<<gCTX, blk256, 0, stream>>>(Pm, Vb, ctxB, SS, SS, HH, HH,
      (long long)SS * SS, (long long)SS * HH, (long long)SS * HH, nullptr);

  // attention out (f32) = ctx @ Wo
  gemm_wmma<float, true><<<gQKV, blk256, 0, stream>>>(ctxB, WoT, ao, HH, HH, HH, UU, 0, 0, 0, nullptr);

  // ln1 = LN(ao + emb), with bf16 aux copy for the next GEMM
  ln_residual_kernel<<<M / 8, blk256, 0, stream>>>(ao, emb, gamma, beta, ln1, ln1B, M);

  // xp = ln1 @ lstm_kernel + bias -> [16384, 2048] f32
  dim3 gXP(M / 128, (4 * UU) / 64, 1);
  gemm_wmma<float, true><<<gXP, blk256, 0, stream>>>(ln1B, lkT, xp, UU, UU, UU, 4 * UU, 0, 0, 0, lb);

  // persistent LSTM scan (TDM h-broadcast + grid barrier per step)
  size_t scan_lds = 64 * 512 * 2 + 16 * 512 * 2 + 4 * 256 * 4 + 256 * 4;
  lstm_scan_kernel<<<32, blk128, scan_lds, stream>>>(xp, lr, hbf, hs, bar);

  // out = LN(hs + ln1)
  ln_residual_kernel<<<M / 8, blk256, 0, stream>>>(hs, ln1, gamma, beta, out, nullptr, M);
}
